// MoE_74105365725744
// MI455X (gfx1250) — matrix operand order repaired and checked by reading, not compiled, not run on hardware
//
#include <hip/hip_runtime.h>
#include <hip/hip_bf16.h>

typedef __bf16 bf16;
typedef __attribute__((ext_vector_type(16))) __bf16 v16bf;
typedef __attribute__((ext_vector_type(8)))  __bf16 v8bf;
typedef __attribute__((ext_vector_type(8)))  float  v8f;

#define BATCH 512
#define TCH   256
#define NE    16
#define NCLS  1000
#define CAPN  128

// ---------------- elementwise f32 -> bf16 ----------------
__global__ void k_cvt_bf16(const float* __restrict__ in, bf16* __restrict__ out, int n) {
  int i = blockIdx.x * blockDim.x + threadIdx.x;
  if (i < n) out[i] = (bf16)in[i];
}

// ---------------- NCHW f32 -> NHWC bf16 (for the network input x) ----------------
__global__ void k_nchw2nhwc(const float* __restrict__ in, bf16* __restrict__ out,
                            int C, int HW, int n) {
  int i = blockIdx.x * blockDim.x + threadIdx.x;  // output-linear: ((b*HW)+p)*C + c
  if (i >= n) return;
  int c = i % C;
  int t = i / C;
  int p = t % HW;
  int b = t / HW;
  out[i] = (bf16)in[((size_t)b * C + c) * HW + p];
}

// ---------------- fold conv-bias + BN into alpha/beta ----------------
__global__ void k_prep_bn(const float* __restrict__ cb, const float* __restrict__ g,
                          const float* __restrict__ bb, const float* __restrict__ m,
                          const float* __restrict__ v, float* __restrict__ alpha,
                          float* __restrict__ beta, int co) {
  int i = blockIdx.x * blockDim.x + threadIdx.x;
  if (i < co) {
    float a = g[i] * rsqrtf(v[i] + 1e-5f);
    alpha[i] = a;
    beta[i]  = (cb[i] - m[i]) * a + bb[i];
  }
}

// ---------------- pack conv weights into WMMA bf16 A-fragment layout ----------------
// K ordering is channel-minor: k = r*CI + ci  (r = 3x3 tap, ci = input channel)
// fragment: wp[((ct*kChunks + kc)*32 + lane)*16 + j]
// K index : kc*32 + j + (lane>=16 ? 16 : 0)
__global__ void k_pack_w(const float* __restrict__ w, bf16* __restrict__ wp,
                         int CI, int kChunks, int n) {
  int idx = blockIdx.x * blockDim.x + threadIdx.x;
  if (idx >= n) return;
  int j    = idx & 15;
  int lane = (idx >> 4) & 31;
  int t    = idx >> 9;
  int kc   = t % kChunks;
  int ct   = t / kChunks;
  int k    = kc * 32 + j + ((lane & 16) ? 16 : 0);
  int co   = ct * 16 + (lane & 15);
  int K    = CI * 9;
  if (k < K) {
    int r  = k / CI;
    int ci = k - r * CI;
    wp[idx] = (bf16)w[((size_t)co * CI + ci) * 9 + r];
  } else {
    wp[idx] = (bf16)0.0f;
  }
}

// ---------------- implicit-GEMM 3x3 pad1 conv + fused BN + ReLU (bf16 WMMA) --------
// Activations are NHWC.  One block: 128-pixel slab x ALL output channels.
// LDS slab [row][x+1][ci] (zero halo, +8-half pad) is filled with per-lane 16B
// GLOBAL_LOAD_ASYNC_TO_LDS_B128 ops (ASYNCcnt), then each lane's B fragment is one
// aligned 32B LDS vector load (16 consecutive ci of one tap) when CI % 32 == 0.
template <int CI, int H, int W, int CO>
__global__ __launch_bounds__(256) void k_conv(
    const bf16* __restrict__ in, const bf16* __restrict__ wp,
    const float* __restrict__ alpha, const float* __restrict__ beta,
    bf16* __restrict__ out) {
  constexpr int K    = CI * 9;
  constexpr int KCH  = (K + 31) / 32;
  constexpr int R    = 128 / W;   // slab rows computed per block
  constexpr int ROWS = R + 2;     // + halo
  constexpr int WP   = W + 2;     // + halo
  constexpr int CIP  = CI + 8;    // pad to break 64-bank stride conflicts
  constexpr bool VEC = (CI % 32) == 0;
  constexpr bool ASY = (CI % 8) == 0;
  __shared__ alignas(32) bf16 slab[ROWS * WP * CIP];

  const int b    = blockIdx.y;
  const int y0   = blockIdx.x * R;
  const int tid  = threadIdx.x;
  const int wave = tid >> 5;
  const int lane = tid & 31;
  const int col  = lane & 15;
  const int hi16 = lane & 16;
  const bf16* __restrict__ inb = in + (size_t)b * H * W * CI;   // NHWC

  if constexpr (ASY) {
    // zero the x-halo columns (x = -1 and x = W) for all rows
    for (int q = tid; q < ROWS * 2; q += 256) {
      int row = q >> 1;
      int xh  = (q & 1) ? (WP - 1) : 0;
      v8bf* d = (v8bf*)(slab + (row * WP + xh) * CIP);
      v8bf z = {};
#pragma unroll
      for (int c8 = 0; c8 < CIP / 8; ++c8) d[c8] = z;
    }
    // interior: async 16B global->LDS copies; OOB rows: zero fill
    constexpr int CH = CI / 8;  // 16B chunks per pixel
    for (int q = tid; q < ROWS * W * CH; q += 256) {
      int c8  = q % CH;
      int pp  = q / CH;
      int x   = pp % W;
      int row = pp / W;
      int yy  = y0 - 1 + row;
      bf16* dst = slab + (row * WP + x + 1) * CIP + c8 * 8;
      if (yy >= 0 && yy < H) {
        unsigned long long ga =
            (unsigned long long)(const void*)(inb + ((size_t)yy * W + x) * CI + c8 * 8);
        unsigned lo = (unsigned)(size_t)(void*)dst;
        asm volatile("global_load_async_to_lds_b128 %0, %1, off"
                     :: "v"(lo), "v"(ga) : "memory");
      } else {
        v8bf z = {};
        *(v8bf*)dst = z;
      }
    }
    asm volatile("s_wait_asynccnt 0x0" ::: "memory");
  } else {
    // generic scalar staging (conv1, CI=3): NHWC source is contiguous per pixel
    for (int p = tid; p < ROWS * WP; p += 256) {
      int row = p / WP;
      int xx  = p - row * WP - 1;
      int yy  = y0 - 1 + row;
      bf16* dst = slab + p * CIP;
      if (xx >= 0 && xx < W && yy >= 0 && yy < H) {
        const bf16* src = inb + ((size_t)yy * W + xx) * CI;
        for (int ci = 0; ci < CI; ++ci) dst[ci] = src[ci];
      } else {
        for (int ci = 0; ci < CI; ++ci) dst[ci] = (bf16)0.0f;
      }
    }
  }
  __syncthreads();

  bf16* __restrict__ outb = out + (size_t)b * H * W * CO;  // NHWC

  for (int t = wave; t < (CO / 16) * 8; t += 8) {
    const int ct = t >> 3;          // co tile
    const int pt = t & 7;           // 16-pixel sub-tile
    const int p  = pt * 16 + col;   // pixel within slab [0,128)
    const int yL = p / W;
    const int x  = p & (W - 1);
    const bf16* __restrict__ wpt = wp + ((size_t)ct * KCH * 32 + lane) * 16;
    v8f acc = {};
#pragma unroll
    for (int kc = 0; kc < KCH; ++kc) {
      v16bf a = *(const v16bf*)(wpt + (size_t)kc * 512);
      v16bf bm = {};
      if constexpr (VEC) {
        const int r   = (kc * 32) / CI;                 // tap, constant per kc
        const int ci0 = (kc * 32) % CI + hi16;          // 16 consecutive channels
        const v8bf* vp = (const v8bf*)(slab + ((yL + r / 3) * WP + (x + r % 3)) * CIP + ci0);
        bm = __builtin_shufflevector(vp[0], vp[1],
                                     0, 1, 2, 3, 4, 5, 6, 7, 8, 9, 10, 11, 12, 13, 14, 15);
      } else {
        const int kbase = kc * 32 + hi16;
#pragma unroll
        for (int j = 0; j < 16; ++j) {
          int k = kbase + j;
          float v = 0.0f;
          if (k < K) {
            int r  = k / CI;
            int ci = k - r * CI;
            v = (float)slab[((yL + r / 3) * WP + (x + r % 3)) * CIP + ci];
          }
          bm[j] = (bf16)v;
        }
      }
      acc = __builtin_amdgcn_wmma_f32_16x16x32_bf16(false, a, false, bm,
                                                    (short)0, acc, false, false);
    }
    // epilogue: 8 consecutive co per lane -> one 16B NHWC store
    const int mo = hi16 ? 8 : 0;
    const int cb = ct * 16 + mo;
    v8bf ov;
#pragma unroll
    for (int vg = 0; vg < 8; ++vg) {
      float o = acc[vg] * alpha[cb + vg] + beta[cb + vg];
      ov[vg] = (bf16)(o > 0.0f ? o : 0.0f);
    }
    *(v8bf*)(outb + ((size_t)(y0 + yL) * W + x) * CO + cb) = ov;
  }
}

// ---------------- 2x2 maxpool (NHWC bf16) ----------------
__global__ void k_maxpool(const bf16* __restrict__ in, bf16* __restrict__ out,
                          int C, int H, int W, int n) {
  int i = blockIdx.x * blockDim.x + threadIdx.x;  // ((b*Ho+oy)*Wo+ox)*C + c
  if (i >= n) return;
  int Ho = H >> 1, Wo = W >> 1;
  int c  = i % C; int t = i / C;
  int ox = t % Wo; t /= Wo;
  int oy = t % Ho; int b = t / Ho;
  const bf16* p = in + (((size_t)b * H + oy * 2) * W + ox * 2) * C + c;
  float m = fmaxf(fmaxf((float)p[0], (float)p[C]),
                  fmaxf((float)p[(size_t)W * C], (float)p[(size_t)W * C + C]));
  out[i] = (bf16)m;
}

// ---------------- global average pool over 256 pixels (NHWC) -> feats f32 ----------
__global__ void k_avgpool(const bf16* __restrict__ in, float* __restrict__ feats) {
  int i = blockIdx.x * blockDim.x + threadIdx.x;  // b*TCH + c
  if (i >= BATCH * TCH) return;
  int c = i % TCH, b = i / TCH;
  const bf16* p = in + (size_t)b * 256 * TCH + c;
  float s = 0.0f;
  for (int j = 0; j < 256; ++j) s += (float)p[(size_t)j * TCH];
  feats[i] = s * (1.0f / 256.0f);
}

// ---------------- pack feats into WMMA A-fragment layout (bf16) ----------------
__global__ void k_pack_feats(const float* __restrict__ feats, bf16* __restrict__ fp, int n) {
  int idx = blockIdx.x * blockDim.x + threadIdx.x;
  if (idx >= n) return;
  int j    = idx & 15;
  int lane = (idx >> 4) & 31;
  int t    = idx >> 9;
  int kc   = t & 7;
  int bt   = t >> 3;
  int row  = bt * 16 + (lane & 15);
  int k    = kc * 32 + j + ((lane & 16) ? 16 : 0);
  fp[idx]  = (bf16)feats[row * TCH + k];
}

// ---------------- per-expert classifier logits (bf16 WMMA GEMM) ----------------
__global__ __launch_bounds__(256) void k_cls_wmma(
    const bf16* __restrict__ fp, const bf16* __restrict__ cw,
    const float* __restrict__ cbias, float* __restrict__ logits) {
  const int wave  = threadIdx.x >> 5;
  const int lane  = threadIdx.x & 31;
  const int nTile = blockIdx.x * 8 + wave;  // 0..999
  const int bTile = blockIdx.y;             // 0..31
  const int ec    = nTile * 16 + (lane & 15);
  const int kHiB  = (lane & 16) ? 16 : 0;
  v8f acc = {};
#pragma unroll
  for (int kc = 0; kc < 8; ++kc) {
    v16bf a = *(const v16bf*)(fp + ((size_t)(bTile * 8 + kc) * 32 + lane) * 16);
    v16bf bmat = *(const v16bf*)(cw + (size_t)ec * TCH + kc * 32 + kHiB);
    acc = __builtin_amdgcn_wmma_f32_16x16x32_bf16(false, a, false, bmat,
                                                  (short)0, acc, false, false);
  }
  const int mo = (lane & 16) ? 8 : 0;
  float bias = cbias[ec];
#pragma unroll
  for (int vg = 0; vg < 8; ++vg) {
    int br = bTile * 16 + mo + vg;
    logits[(size_t)br * (NE * NCLS) + ec] = acc[vg] + bias;
  }
}

// ---------------- gate MLP ----------------
__global__ void k_gate1(const float* __restrict__ feats, const float* __restrict__ w1,
                        const float* __restrict__ b1, float* __restrict__ g) {
  int i = blockIdx.x * blockDim.x + threadIdx.x;  // b*128+o
  if (i >= BATCH * 128) return;
  int o = i & 127, b = i >> 7;
  const float* f = feats + b * TCH;
  const float* w = w1 + o * TCH;
  float s = b1[o];
  for (int d = 0; d < TCH; ++d) s += f[d] * w[d];
  g[i] = s > 0.0f ? s : 0.0f;
}

__global__ void k_gate2(const float* __restrict__ g, const float* __restrict__ w2,
                        const float* __restrict__ b2, float* __restrict__ gl) {
  int i = blockIdx.x * blockDim.x + threadIdx.x;  // b*16+e
  if (i >= BATCH * NE) return;
  int e = i & 15, b = i >> 4;
  const float* gg = g + b * 128;
  const float* w  = w2 + e * 128;
  float s = b2[e];
  for (int d = 0; d < 128; ++d) s += gg[d] * w[d];
  gl[i] = s;
}

// ---------------- softmax entropy -> confidence -> routing scores ----------------
__global__ __launch_bounds__(256) void k_entropy_route(
    const float* __restrict__ logits, const float* __restrict__ gl,
    float* __restrict__ routing) {
  const int be = blockIdx.x;                 // b*16+e
  const float* L = logits + (size_t)be * NCLS;
  __shared__ float red[256];
  int tid = threadIdx.x;
  float m = -3.4e38f;
  for (int c = tid; c < NCLS; c += 256) m = fmaxf(m, L[c]);
  red[tid] = m; __syncthreads();
  for (int s = 128; s > 0; s >>= 1) { if (tid < s) red[tid] = fmaxf(red[tid], red[tid + s]); __syncthreads(); }
  m = red[0]; __syncthreads();
  float se = 0.0f, sx = 0.0f;
  for (int c = tid; c < NCLS; c += 256) {
    float x = L[c] - m;
    float e = __expf(x);
    se += e; sx += x * e;
  }
  red[tid] = se; __syncthreads();
  for (int s = 128; s > 0; s >>= 1) { if (tid < s) red[tid] += red[tid + s]; __syncthreads(); }
  se = red[0]; __syncthreads();
  red[tid] = sx; __syncthreads();
  for (int s = 128; s > 0; s >>= 1) { if (tid < s) red[tid] += red[tid + s]; __syncthreads(); }
  sx = red[0];
  if (tid == 0) {
    float ent = __logf(se) - sx / se;      // = -sum p*log p
    routing[be] = gl[be] * (-ent);         // gate * confidence
  }
}

// ---------------- capacity top-128 per expert (rank counting, top_k tie rule) ------
__global__ __launch_bounds__(512) void k_capacity(const float* __restrict__ routing,
                                                  int* __restrict__ D1) {
  int e = blockIdx.x;
  int b = threadIdx.x;
  __shared__ float s[512];
  s[b] = routing[b * NE + e];
  __syncthreads();
  float v = s[b];
  int cnt = 0;
  for (int i = 0; i < 512; ++i) {
    float u = s[i];
    cnt += (u > v) || (u == v && i < b);
  }
  D1[b * NE + e] = (cnt < CAPN) ? 1 : 0;
}

// ---------------- per-token top-2 among admitted (masked) ----------------
__global__ void k_token_topk(const float* __restrict__ routing, const int* __restrict__ D1,
                             int* __restrict__ sel, float* __restrict__ Dout) {
  int b = blockIdx.x * blockDim.x + threadIdx.x;
  if (b >= BATCH) return;
  float best1 = -3.4e38f; int i1 = 0;
  for (int e = 0; e < NE; ++e) {
    float v = D1[b * NE + e] ? routing[b * NE + e] : -1e9f;
    if (v > best1) { best1 = v; i1 = e; }
  }
  float best2 = -3.4e38f; int i2 = 0;
  for (int e = 0; e < NE; ++e) {
    if (e == i1) continue;
    float v = D1[b * NE + e] ? routing[b * NE + e] : -1e9f;
    if (v > best2) { best2 = v; i2 = e; }
  }
  sel[b * 2] = i1; sel[b * 2 + 1] = i2;
  for (int e = 0; e < NE; ++e)
    Dout[b * NE + e] = (e == i1 || e == i2) ? 1.0f : 0.0f;
}

// ---------------- weighted combine: top_k always returns K indices -> norm = 2 -----
__global__ void k_combine(const float* __restrict__ logits, const float* __restrict__ routing,
                          const int* __restrict__ sel, float* __restrict__ out) {
  int b = blockIdx.x;
  int c = blockIdx.y * 256 + threadIdx.x;
  if (c >= NCLS) return;
  int e1 = sel[b * 2], e2 = sel[b * 2 + 1];
  float w1 = routing[b * NE + e1], w2 = routing[b * NE + e2];
  float v = w1 * logits[((size_t)b * NE + e1) * NCLS + c] +
            w2 * logits[((size_t)b * NE + e2) * NCLS + c];
  out[(size_t)b * NCLS + c] = v * 0.5f;
}

extern "C" void kernel_launch(void* const* d_in, const int* in_sizes, int n_in,
                              void* d_out, int out_size, void* d_ws, size_t ws_size,
                              hipStream_t stream) {
  const float* x = (const float*)d_in[0];
  const float* convw[4] = {(const float*)d_in[1], (const float*)d_in[7],
                           (const float*)d_in[13], (const float*)d_in[19]};
  const float* convb[4] = {(const float*)d_in[2], (const float*)d_in[8],
                           (const float*)d_in[14], (const float*)d_in[20]};
  const float* bng[4] = {(const float*)d_in[3], (const float*)d_in[9],
                         (const float*)d_in[15], (const float*)d_in[21]};
  const float* bnb[4] = {(const float*)d_in[4], (const float*)d_in[10],
                         (const float*)d_in[16], (const float*)d_in[22]};
  const float* bnm[4] = {(const float*)d_in[5], (const float*)d_in[11],
                         (const float*)d_in[17], (const float*)d_in[23]};
  const float* bnv[4] = {(const float*)d_in[6], (const float*)d_in[12],
                         (const float*)d_in[18], (const float*)d_in[24]};
  const float* gate_w1 = (const float*)d_in[25];
  const float* gate_b1 = (const float*)d_in[26];
  const float* gate_w2 = (const float*)d_in[27];
  const float* gate_b2 = (const float*)d_in[28];
  const float* cls_w   = (const float*)d_in[29];
  const float* cls_b   = (const float*)d_in[30];

  const int Ci[4] = {3, 32, 64, 128};
  const int Co[4] = {32, 64, 128, 256};

  char* ws = (char*)d_ws;
  size_t off = 0;
  auto alloc = [&](size_t bytes) -> char* {
    char* p = ws + off;
    off += (bytes + 255) & ~(size_t)255;
    return p;
  };

  bf16* xbf  = (bf16*)alloc((size_t)BATCH * 3 * 64 * 64 * 2);
  bf16* bufA = (bf16*)alloc((size_t)BATCH * 32 * 64 * 64 * 2);   // h1 / h3
  bf16* bufB = (bf16*)alloc((size_t)BATCH * 64 * 64 * 64 * 2);   // h2 / h4
  bf16* bufP = (bf16*)alloc((size_t)BATCH * 64 * 32 * 32 * 2);   // p2 / p4
  const int kChunks[4] = {1, 9, 18, 36};
  bf16* wp[4];
  float *alpha[4], *beta[4];
  for (int l = 0; l < 4; ++l) {
    wp[l]    = (bf16*)alloc((size_t)(Co[l] / 16) * kChunks[l] * 512 * 2);
    alpha[l] = (float*)alloc((size_t)Co[l] * 4);
    beta[l]  = (float*)alloc((size_t)Co[l] * 4);
  }
  float* feats  = (float*)alloc((size_t)BATCH * TCH * 4);
  bf16* fpack   = (bf16*)alloc((size_t)32 * 8 * 512 * 2);
  bf16* cwbf    = (bf16*)alloc((size_t)NE * NCLS * TCH * 2);
  float* logits = (float*)alloc((size_t)BATCH * NE * NCLS * 4);
  float* g1buf  = (float*)alloc((size_t)BATCH * 128 * 4);
  float* glog   = (float*)alloc((size_t)BATCH * NE * 4);
  int* D1       = (int*)alloc((size_t)BATCH * NE * 4);
  int* sel      = (int*)alloc((size_t)BATCH * 2 * 4);

  float* out_final   = (float*)d_out;                         // 512*1000
  float* out_routing = (float*)d_out + (size_t)BATCH * NCLS;  // 512*16
  float* out_D       = out_routing + (size_t)BATCH * NE;      // 512*16

  // ---- prep: x -> NHWC bf16, fold BN, pack weights ----
  {
    int n = BATCH * 3 * 64 * 64;
    k_nchw2nhwc<<<(n + 255) / 256, 256, 0, stream>>>(x, xbf, 3, 64 * 64, n);
  }
  for (int l = 0; l < 4; ++l) {
    k_prep_bn<<<(Co[l] + 63) / 64, 64, 0, stream>>>(convb[l], bng[l], bnb[l], bnm[l],
                                                    bnv[l], alpha[l], beta[l], Co[l]);
    int n = (Co[l] / 16) * kChunks[l] * 512;
    k_pack_w<<<(n + 255) / 256, 256, 0, stream>>>(convw[l], wp[l], Ci[l], kChunks[l], n);
  }

  // ---- trunk (NHWC) ----
  k_conv<3, 64, 64, 32><<<dim3(32, BATCH), 256, 0, stream>>>(xbf, wp[0], alpha[0], beta[0], bufA);
  k_conv<32, 64, 64, 64><<<dim3(32, BATCH), 256, 0, stream>>>(bufA, wp[1], alpha[1], beta[1], bufB);
  {
    int n = BATCH * 64 * 32 * 32;
    k_maxpool<<<(n + 255) / 256, 256, 0, stream>>>(bufB, bufP, 64, 64, 64, n);
  }
  k_conv<64, 32, 32, 128><<<dim3(8, BATCH), 256, 0, stream>>>(bufP, wp[2], alpha[2], beta[2], bufA);
  k_conv<128, 32, 32, 256><<<dim3(8, BATCH), 256, 0, stream>>>(bufA, wp[3], alpha[3], beta[3], bufB);
  {
    int n = BATCH * 256 * 16 * 16;
    k_maxpool<<<(n + 255) / 256, 256, 0, stream>>>(bufB, bufP, 256, 32, 32, n);
  }
  {
    int n = BATCH * TCH;
    k_avgpool<<<(n + 255) / 256, 256, 0, stream>>>(bufP, feats);
  }

  // ---- per-expert classifier logits (WMMA GEMM) ----
  {
    int n = NE * NCLS * TCH;
    k_cvt_bf16<<<(n + 255) / 256, 256, 0, stream>>>(cls_w, cwbf, n);
  }
  {
    int n = 32 * 8 * 512;
    k_pack_feats<<<(n + 255) / 256, 256, 0, stream>>>(feats, fpack, n);
  }
  k_cls_wmma<<<dim3(125, 32), 256, 0, stream>>>(fpack, cwbf, cls_b, logits);

  // ---- gate ----
  k_gate1<<<(BATCH * 128 + 255) / 256, 256, 0, stream>>>(feats, gate_w1, gate_b1, g1buf);
  k_gate2<<<(BATCH * NE + 255) / 256, 256, 0, stream>>>(g1buf, gate_w2, gate_b2, glog);

  // ---- entropy confidence + routing scores (written straight to d_out slot) ----
  k_entropy_route<<<BATCH * NE, 256, 0, stream>>>(logits, glog, out_routing);

  // ---- capacity + token top-k + combine ----
  k_capacity<<<NE, 512, 0, stream>>>(out_routing, D1);
  k_token_topk<<<(BATCH + 255) / 256, 256, 0, stream>>>(out_routing, D1, sel, out_D);
  k_combine<<<dim3(BATCH, (NCLS + 255) / 256), 256, 0, stream>>>(logits, out_routing, sel,
                                                                 out_final);
}
